// GraphTransformerModel_13486197309562
// MI455X (gfx1250) — matrix operand order repaired and checked by reading, not compiled, not run on hardware
//
#include <hip/hip_runtime.h>
#include <hip/hip_bf16.h>

// ---------------------------------------------------------------------------
// GraphTransformer (3x TransformerConv + LN) for MI455X (gfx1250, wave32).
// GEMMs via v_wmma_f32_16x16x32_bf16 with fully-contiguous fragment loads
// (A row-major, B pre-transposed at conversion time -> b128 loads, no packing).
// Edge phase: wave-per-edge float4 gathers (q+k+v = 123MB < 192MB L2) +
// native f32 atomics.
// ---------------------------------------------------------------------------

typedef __attribute__((ext_vector_type(16))) __bf16 v16bf;
typedef __attribute__((ext_vector_type(8)))  __bf16 v8bf;
typedef __attribute__((ext_vector_type(8)))  float  v8f;

#define D_IN   64
#define HID    128
#define NHEAD  4
#define NLAYER 3
#define QKVDIM 512   // NHEAD * HID

// ---------------- elementwise helpers ----------------

__global__ void f32_to_bf16_kernel(const float* __restrict__ in,
                                   __bf16* __restrict__ out, long long n) {
  long long i = (long long)blockIdx.x * blockDim.x + threadIdx.x;
  if (i < n) out[i] = (__bf16)in[i];
}

// out[col*K + k] = (bf16) in[k*Ncol + col]   (weight transpose + convert)
__global__ void f32_to_bf16_T_kernel(const float* __restrict__ in,
                                     __bf16* __restrict__ out, int K, int Ncol) {
  int idx = blockIdx.x * blockDim.x + threadIdx.x;
  if (idx >= K * Ncol) return;
  int col = idx / K;
  int k   = idx - col * K;
  out[idx] = (__bf16)in[k * Ncol + col];
}

__global__ void fill_f32_kernel(float* __restrict__ p, float v, long long n) {
  long long i = (long long)blockIdx.x * blockDim.x + threadIdx.x;
  if (i < n) p[i] = v;
}

__device__ inline float atomicMaxF(float* addr, float val) {
  unsigned int* ua = (unsigned int*)addr;
  unsigned int old = *ua;
  while (true) {
    float f = __uint_as_float(old);
    if (f >= val) break;
    unsigned int assumed = old;
    old = atomicCAS(ua, assumed, __float_as_uint(val));
    if (old == assumed) break;
  }
  return __uint_as_float(old);
}

// ---------------- WMMA GEMM ----------------
// C[M x Ncol] = A[M x K](bf16, row-major) @ Bt[Ncol x K](bf16, transposed) + bias
// One wave computes a 16x32 output (two 16x16 WMMA tiles sharing the A frag).
// Fragment layouts used here (wave32; A/B in contiguous k order, not the ISA table order):
//   A: lane row = lane%16; half = lane/16;
//      elems 0..15 -> K = k0 + half*16 + (0..15)    (contiguous, 32B load)
//   B: lane col = lane%16; elems 0..15 -> K = k0 + half*16 + (0..15)
//      contiguous in transposed storage -> single 32B load
//   D: vgpr j -> row = j + 8*half, col = lane%16

#define GEMM_WAVES 4

template <int K>
__global__ void gemm_bf16_wmma(const __bf16* __restrict__ A,
                               const __bf16* __restrict__ Bt,
                               const float*  __restrict__ bias,
                               float*        __restrict__ Cf,
                               __bf16*       __restrict__ Cbf,
                               int M, int Ncol, int ldc) {
  const int lane = threadIdx.x & 31;
  const int wave = threadIdx.x >> 5;
  const int tm   = blockIdx.x;
  const int n0   = (blockIdx.y * GEMM_WAVES + wave) * 32;
  if (n0 >= Ncol) return;
  const int r    = lane & 15;
  const int half = lane >> 4;
  const int rowA = tm * 16 + r;
  const int col0 = n0 + r;
  const int col1 = n0 + 16 + r;

  const __bf16* __restrict__ aRow = A + (size_t)rowA * K;
  const __bf16* __restrict__ b0p  = Bt + (size_t)col0 * K;
  const __bf16* __restrict__ b1p  = Bt + (size_t)col1 * K;

  v8f acc0 = {}, acc1 = {};
#pragma unroll
  for (int k0 = 0; k0 < K; k0 += 32) {
    const v16bf a = *(const v16bf*)(aRow + k0 + half * 16);
    const v16bf b0 = *(const v16bf*)(b0p + k0 + half * 16);
    const v16bf b1 = *(const v16bf*)(b1p + k0 + half * 16);
    acc0 = __builtin_amdgcn_wmma_f32_16x16x32_bf16(false, a, false, b0,
                                                   (short)0, acc0, false, false);
    acc1 = __builtin_amdgcn_wmma_f32_16x16x32_bf16(false, a, false, b1,
                                                   (short)0, acc1, false, false);
  }

  const float bb0 = bias ? bias[col0] : 0.0f;
  const float bb1 = bias ? bias[col1] : 0.0f;
#pragma unroll
  for (int j = 0; j < 8; ++j) {
    const int gr = tm * 16 + half * 8 + j;
    const float v0 = acc0[j] + bb0;
    const float v1 = acc1[j] + bb1;
    Cf[(size_t)gr * ldc + col0] = v0;
    Cf[(size_t)gr * ldc + col1] = v1;
    if (Cbf) {
      Cbf[(size_t)gr * ldc + col0] = (__bf16)v0;
      Cbf[(size_t)gr * ldc + col1] = (__bf16)v1;
    }
  }
}

// ---------------- edge phase ----------------

// wave per edge: logits[e,h] = scale * dot(q[dst,h,:], k[src,h,:]); segment-max
__global__ void edge_logits_kernel(const long long* __restrict__ src,
                                   const long long* __restrict__ dst,
                                   const float* __restrict__ q,
                                   const float* __restrict__ k,
                                   float* __restrict__ logits,
                                   float* __restrict__ mmax, int E) {
  const int lane = threadIdx.x & 31;
  const int e = (int)(((long long)blockIdx.x * blockDim.x + threadIdx.x) >> 5);
  if (e >= E) return;
  const long long s = src[e];
  const long long d = dst[e];
  const float scale = 0.08838834764831845f;  // 1/sqrt(128)
#pragma unroll
  for (int h = 0; h < NHEAD; ++h) {
    const float4 qv = reinterpret_cast<const float4*>(q + d * QKVDIM + h * HID)[lane];
    const float4 kv = reinterpret_cast<const float4*>(k + s * QKVDIM + h * HID)[lane];
    float p = qv.x * kv.x + qv.y * kv.y + qv.z * kv.z + qv.w * kv.w;
#pragma unroll
    for (int off = 16; off >= 1; off >>= 1) p += __shfl_xor(p, off, 32);
    if (lane == 0) {
      const float lg = p * scale;
      logits[e * NHEAD + h] = lg;
      atomicMaxF(&mmax[d * NHEAD + h], lg);
    }
  }
}

// thread per (edge,head): ex = exp(logit - m[dst]); segment-sum; logits <- ex
__global__ void edge_exp_kernel(const long long* __restrict__ dst,
                                float* __restrict__ logits,
                                const float* __restrict__ mmax,
                                float* __restrict__ ssum, int E) {
  const int i = blockIdx.x * blockDim.x + threadIdx.x;
  if (i >= E * NHEAD) return;
  const int e = i >> 2, h = i & 3;
  const long long d = dst[e];
  const float ex = __expf(logits[i] - mmax[d * NHEAD + h]);
  logits[i] = ex;
  atomicAdd(&ssum[d * NHEAD + h], ex);
}

// wave per edge: agg[dst,h,:] += (ex / (sum+eps)) * v[src,h,:]
__global__ void message_kernel(const long long* __restrict__ src,
                               const long long* __restrict__ dst,
                               const float* __restrict__ v,
                               const float* __restrict__ logits,
                               const float* __restrict__ ssum,
                               float* __restrict__ agg, int E) {
  const int lane = threadIdx.x & 31;
  const int e = (int)(((long long)blockIdx.x * blockDim.x + threadIdx.x) >> 5);
  if (e >= E) return;
  const long long s = src[e], d = dst[e];
#pragma unroll
  for (int h = 0; h < NHEAD; ++h) {
    const float alpha = logits[e * NHEAD + h] / (ssum[d * NHEAD + h] + 1e-16f);
    const float4 vv = reinterpret_cast<const float4*>(v + s * QKVDIM + h * HID)[lane];
    float* o = agg + d * QKVDIM + h * HID + lane * 4;
    atomicAdd(o + 0, alpha * vv.x);
    atomicAdd(o + 1, alpha * vv.y);
    atomicAdd(o + 2, alpha * vv.z);
    atomicAdd(o + 3, alpha * vv.w);
  }
}

// wave per node: mean over heads + skip + residual + LayerNorm (shuffle reduce)
__global__ void finalize_ln_kernel(const float* __restrict__ hin,
                                   const float* __restrict__ agg,
                                   const float* __restrict__ skip,
                                   const float* __restrict__ gamma,
                                   const float* __restrict__ beta,
                                   float* __restrict__ hout,
                                   __bf16* __restrict__ hbf, int N) {
  const int lane = threadIdx.x & 31;
  const int n = (int)(((long long)blockIdx.x * blockDim.x + threadIdx.x) >> 5);
  if (n >= N) return;
  float vals[4];
#pragma unroll
  for (int j = 0; j < 4; ++j) {
    const int c = lane * 4 + j;
    const float mh = 0.25f * (agg[n * QKVDIM + 0 * HID + c] +
                              agg[n * QKVDIM + 1 * HID + c] +
                              agg[n * QKVDIM + 2 * HID + c] +
                              agg[n * QKVDIM + 3 * HID + c]);
    vals[j] = hin[n * HID + c] + mh + skip[n * HID + c];
  }
  float su = vals[0] + vals[1] + vals[2] + vals[3];
#pragma unroll
  for (int off = 16; off >= 1; off >>= 1) su += __shfl_xor(su, off, 32);
  const float mu = su * (1.0f / 128.0f);
  float sv = 0.f;
#pragma unroll
  for (int j = 0; j < 4; ++j) { const float t = vals[j] - mu; sv += t * t; }
#pragma unroll
  for (int off = 16; off >= 1; off >>= 1) sv += __shfl_xor(sv, off, 32);
  const float rstd = rsqrtf(sv * (1.0f / 128.0f) + 1e-5f);
#pragma unroll
  for (int j = 0; j < 4; ++j) {
    const int c = lane * 4 + j;
    const float o = (vals[j] - mu) * rstd * gamma[c] + beta[c];
    hout[n * HID + c] = o;
    if (hbf) hbf[n * HID + c] = (__bf16)o;
  }
}

// ---------------------------------------------------------------------------

extern "C" void kernel_launch(void* const* d_in, const int* in_sizes, int n_in,
                              void* d_out, int out_size, void* d_ws, size_t ws_size,
                              hipStream_t stream) {
  const float*     x      = (const float*)d_in[0];
  const long long* ei     = (const long long*)d_in[1];
  const float*     Win    = (const float*)d_in[2];
  const float*     bin_   = (const float*)d_in[3];
  const float*     Wq     = (const float*)d_in[4];
  const float*     bq     = (const float*)d_in[5];
  const float*     Wk     = (const float*)d_in[6];
  const float*     bk     = (const float*)d_in[7];
  const float*     Wv     = (const float*)d_in[8];
  const float*     bv     = (const float*)d_in[9];
  const float*     Wskip  = (const float*)d_in[10];
  const float*     bskip  = (const float*)d_in[11];
  const float*     gamma  = (const float*)d_in[12];
  const float*     beta   = (const float*)d_in[13];

  const int N = in_sizes[0] / D_IN;
  const int E = in_sizes[1] / 2;
  const long long* src = ei;
  const long long* dst = ei + E;

  // ---- workspace carve-out ----
  char* ws = (char*)d_ws;
  size_t off = 0;
  auto carve = [&](size_t bytes) -> void* {
    void* p = ws + off;
    off = (off + bytes + 255) & ~(size_t)255;
    return p;
  };
  float*  h_f32  = (float*)carve((size_t)N * HID * 4);
  __bf16* h_bf   = (__bf16*)carve((size_t)N * HID * 2);
  __bf16* x_bf   = (__bf16*)carve((size_t)N * D_IN * 2);
  __bf16* w_bf   = (__bf16*)carve((size_t)HID * QKVDIM * 2);   // reused per GEMM
  float*  qbuf   = (float*)carve((size_t)N * QKVDIM * 4);      // doubles as agg
  float*  kbuf   = (float*)carve((size_t)N * QKVDIM * 4);
  float*  vbuf   = (float*)carve((size_t)N * QKVDIM * 4);
  float*  skip   = (float*)carve((size_t)N * HID * 4);
  float*  logits = (float*)carve((size_t)E * NHEAD * 4);
  float*  mmax   = (float*)carve((size_t)N * NHEAD * 4);
  float*  ssum   = (float*)carve((size_t)N * NHEAD * 4);
  (void)ws_size;

  auto conv = [&](const float* in, __bf16* out, long long n) {
    f32_to_bf16_kernel<<<(unsigned)((n + 255) / 256), 256, 0, stream>>>(in, out, n);
  };
  auto convT = [&](const float* in, __bf16* out, int K, int Ncol) {
    long long n = (long long)K * Ncol;
    f32_to_bf16_T_kernel<<<(unsigned)((n + 255) / 256), 256, 0, stream>>>(in, out, K, Ncol);
  };
  auto gemm = [&](const __bf16* A, const __bf16* Bt, const float* bias,
                  float* Cf, __bf16* Cbf, int M, int K, int Ncol, int ldc) {
    dim3 grid(M / 16, (Ncol / 32 + GEMM_WAVES - 1) / GEMM_WAVES);
    if (K == 64)
      gemm_bf16_wmma<64><<<grid, 32 * GEMM_WAVES, 0, stream>>>(A, Bt, bias, Cf, Cbf,
                                                               M, Ncol, ldc);
    else
      gemm_bf16_wmma<128><<<grid, 32 * GEMM_WAVES, 0, stream>>>(A, Bt, bias, Cf, Cbf,
                                                                M, Ncol, ldc);
  };
  auto fill = [&](float* p, float v, long long n) {
    fill_f32_kernel<<<(unsigned)((n + 255) / 256), 256, 0, stream>>>(p, v, n);
  };

  const int edgeWaveBlocks = (E * 32 + 255) / 256;
  const int nodeWaveBlocks = (N * 32 + 255) / 256;

  // ---- input projection: h = x @ Win + bin ----
  conv(x, x_bf, (long long)N * D_IN);
  convT(Win, w_bf, D_IN, HID);
  gemm(x_bf, w_bf, bin_, h_f32, h_bf, N, D_IN, HID, HID);

  // ---- layers ----
  for (int l = 0; l < NLAYER; ++l) {
    const size_t wqkv_off = (size_t)l * HID * QKVDIM;
    const size_t wsk_off  = (size_t)l * HID * HID;

    convT(Wq + wqkv_off, w_bf, HID, QKVDIM);
    gemm(h_bf, w_bf, bq + (size_t)l * QKVDIM, qbuf, nullptr, N, HID, QKVDIM, QKVDIM);

    convT(Wk + wqkv_off, w_bf, HID, QKVDIM);
    gemm(h_bf, w_bf, bk + (size_t)l * QKVDIM, kbuf, nullptr, N, HID, QKVDIM, QKVDIM);

    convT(Wv + wqkv_off, w_bf, HID, QKVDIM);
    gemm(h_bf, w_bf, bv + (size_t)l * QKVDIM, vbuf, nullptr, N, HID, QKVDIM, QKVDIM);

    convT(Wskip + wsk_off, w_bf, HID, HID);
    gemm(h_bf, w_bf, bskip + (size_t)l * HID, skip, nullptr, N, HID, HID, HID);

    fill(mmax, -3.4e38f, (long long)N * NHEAD);
    fill(ssum, 0.0f, (long long)N * NHEAD);

    edge_logits_kernel<<<edgeWaveBlocks, 256, 0, stream>>>(src, dst, qbuf, kbuf,
                                                           logits, mmax, E);
    edge_exp_kernel<<<(E * NHEAD + 255) / 256, 256, 0, stream>>>(dst, logits, mmax, ssum, E);

    float* agg = qbuf;  // q no longer needed; reuse as aggregation buffer
    fill(agg, 0.0f, (long long)N * QKVDIM);
    message_kernel<<<edgeWaveBlocks, 256, 0, stream>>>(src, dst, vbuf, logits, ssum, agg, E);

    const bool last = (l == NLAYER - 1);
    finalize_ln_kernel<<<nodeWaveBlocks, 256, 0, stream>>>(
        h_f32, agg, skip, gamma + (size_t)l * HID, beta + (size_t)l * HID,
        last ? (float*)d_out : h_f32, last ? nullptr : h_bf, N);
  }
}
